// MaskedTransformerCrossEncoder_28991029248248
// MI455X (gfx1250) — matrix operand order repaired and checked by reading, not compiled, not run on hardware
//
#include <hip/hip_runtime.h>

// ---------------------------------------------------------------------------
// MaskedTransformerCrossEncoder forward for MI455X (gfx1250, wave32, WMMA).
// All matmuls via v_wmma_f32_16x16x32_f16. All GEMM operands stored f16 in
// memory (weights pre-converted once; activations written f16 by producers),
// f32 accumulate, f32 residual stream.
// ---------------------------------------------------------------------------

namespace {
constexpr int kN    = 2048;
constexpr int kD    = 256;
constexpr int kDff  = 1024;
constexpr int kH    = 8;
constexpr int kHd   = 32;
constexpr int kL    = 3;
constexpr long long kND   = (long long)kN * kD;     // 524288
constexpr long long kNDff = (long long)kN * kDff;   // 2097152
constexpr long long kNN   = (long long)kN * kN;     // 4194304
constexpr float kLnEps = 1e-5f;
}

typedef __attribute__((ext_vector_type(16))) _Float16 v16h;
typedef __attribute__((ext_vector_type(8)))  _Float16 v8h;
typedef __attribute__((ext_vector_type(8)))  float    v8f;

// A-fragment (16x32 f16, ISA 7.12.2): lanes 0-15 hold K {0..7, 16..23},
// lanes 16-31 hold K {8..15, 24..31} of their row.  kb = (lane&16) ? 8 : 0.
__device__ __forceinline__ v16h frag_load_a(const _Float16* __restrict__ p, int kb) {
  v8h lo = *reinterpret_cast<const v8h*>(p + kb);
  v8h hi = *reinterpret_cast<const v8h*>(p + kb + 16);
  return __builtin_shufflevector(lo, hi, 0, 1, 2, 3, 4, 5, 6, 7,
                                 8, 9, 10, 11, 12, 13, 14, 15);
}

// B-fragment (32x16 f16): same K order as A: lanes 0-15 hold K {0..7, 16..23},
// lanes 16-31 hold K {8..15, 24..31} of their column.
// kb = (lane&16) ? 8 : 0.  We feed it rows of W (columns of B = W^T).
__device__ __forceinline__ v16h frag_load_b(const _Float16* __restrict__ p, int kb) {
  v8h lo = *reinterpret_cast<const v8h*>(p + kb);
  v8h hi = *reinterpret_cast<const v8h*>(p + kb + 16);
  return __builtin_shufflevector(lo, hi, 0, 1, 2, 3, 4, 5, 6, 7,
                                 8, 9, 10, 11, 12, 13, 14, 15);
}

__device__ __forceinline__ v8f wmma_step(v16h a, v16h b, v8f c) {
  return __builtin_amdgcn_wmma_f32_16x16x32_f16(false, a, false, b,
                                                (short)0, c, false, false);
}

// C[m,n] = alpha * sum_k A[m,k] * W[n,k] (+bias[n]) (+relu), C f32 or f16,
// optionally transposed store.  Per-wave tile 32x32 = 2x2 WMMA tiles (4 wmma
// per 4 fragment loads).  8 waves/block arranged (8/nWaveN) x nWaveN:
//   nWaveN=2 -> 128x64 block tile (generic), nWaveN=1 -> 256x32 (for Nn=32).
// blockIdx.z walks heads via aZ/bZ/cZ element offsets.
__global__ __launch_bounds__(256)
void gemm_wmma_f16(const _Float16* __restrict__ A, const _Float16* __restrict__ W,
                   const float* __restrict__ bias, void* __restrict__ Cv,
                   int M, int Nn, int Kd, int lda, int ldb, int ldc,
                   long long aZ, long long bZ, long long cZ,
                   float alpha, int relu, int storeT, int outHalf, int nWaveN) {
  const int lane = threadIdx.x & 31;
  const int wave = threadIdx.x >> 5;                 // 0..7
  const int wm = (nWaveN == 2) ? (wave >> 1) : wave;
  const int wn = (nWaveN == 2) ? (wave & 1) : 0;
  const int nWaveM = 8 / nWaveN;
  const int row0 = (blockIdx.y * nWaveM + wm) * 32;
  const int col0 = (blockIdx.x * nWaveN + wn) * 32;

  A += (long long)blockIdx.z * aZ;
  W += (long long)blockIdx.z * bZ;
  float*    Cf = (float*)Cv    + (long long)blockIdx.z * cZ;
  _Float16* Ch = (_Float16*)Cv + (long long)blockIdx.z * cZ;

  const int r   = lane & 15;
  const int kba = (lane & 16) ? 8 : 0;    // A: interleaved 8-chunks
  const int kbb = (lane & 16) ? 8 : 0;    // B: interleaved 8-chunks
  const _Float16* a0 = A + (long long)(row0 + r) * lda;
  const _Float16* a1 = A + (long long)(row0 + 16 + r) * lda;
  const _Float16* b0 = W + (long long)(col0 + r) * ldb;
  const _Float16* b1 = W + (long long)(col0 + 16 + r) * ldb;

  v8f acc00 = {0.f, 0.f, 0.f, 0.f, 0.f, 0.f, 0.f, 0.f};
  v8f acc01 = acc00, acc10 = acc00, acc11 = acc00;

#pragma unroll 2
  for (int k = 0; k < Kd; k += 32) {
    v16h fa0 = frag_load_a(a0 + k, kba);
    v16h fa1 = frag_load_a(a1 + k, kba);
    v16h fb0 = frag_load_b(b0 + k, kbb);
    v16h fb1 = frag_load_b(b1 + k, kbb);
    acc00 = wmma_step(fa0, fb0, acc00);
    acc01 = wmma_step(fa0, fb1, acc01);
    acc10 = wmma_step(fa1, fb0, acc10);
    acc11 = wmma_step(fa1, fb1, acc11);
  }

  // C/D layout: VGPR v, lane l -> row = v + 8*(l/16), col = l%16 (ISA 7.12.2)
  const int rsub = (lane & 16) ? 8 : 0;
  const int csub = lane & 15;
#pragma unroll
  for (int t = 0; t < 4; ++t) {
    const v8f acc = (t == 0) ? acc00 : (t == 1) ? acc01 : (t == 2) ? acc10 : acc11;
    const int rbase = row0 + ((t >> 1) ? 16 : 0) + rsub;
    const int ccol  = col0 + ((t & 1) ? 16 : 0) + csub;
    const float bv = bias ? bias[ccol] : 0.0f;
#pragma unroll
    for (int v = 0; v < 8; ++v) {
      float x = acc[v] * alpha + bv;
      if (relu) x = fmaxf(x, 0.0f);
      const long long idx = storeT ? ((long long)ccol * ldc + (rbase + v))
                                   : ((long long)(rbase + v) * ldc + ccol);
      if (outHalf) Ch[idx] = (_Float16)x; else Cf[idx] = x;
    }
  }
}

// In-place masked softmax over f16 rows of S[h][row][0..N).  Radius mask from
// xyz (d^2 >= r2 -> -inf); xyz==nullptr disables masking (cross-attention).
// One block (256 thr) per (row, head); 8 columns per thread in registers.
__global__ __launch_bounds__(256)
void masked_softmax(_Float16* __restrict__ S, const float* __restrict__ xyz,
                    float r2) {
  const int row = blockIdx.x;
  const int h   = blockIdx.y;
  _Float16* s = S + ((long long)h * kN + row) * kN;

  __shared__ float red[8];
  const int lane = threadIdx.x & 31;
  const int wave = threadIdx.x >> 5;

  float mx = 0.f, my = 0.f, mz = 0.f;
  const bool use_mask = (xyz != nullptr);
  if (use_mask) { mx = xyz[row * 3]; my = xyz[row * 3 + 1]; mz = xyz[row * 3 + 2]; }

  float vals[8];
  float vmax = -3.0e38f;
#pragma unroll
  for (int j = 0; j < 8; ++j) {
    const int n = threadIdx.x + j * 256;
    float v = (float)s[n];
    if (use_mask) {
      const float dx = mx - xyz[n * 3];
      const float dy = my - xyz[n * 3 + 1];
      const float dz = mz - xyz[n * 3 + 2];
      if (dx * dx + dy * dy + dz * dz >= r2) v = -3.0e38f;
    }
    vals[j] = v;
    vmax = fmaxf(vmax, v);
  }
#pragma unroll
  for (int off = 16; off > 0; off >>= 1) vmax = fmaxf(vmax, __shfl_xor(vmax, off, 32));
  if (lane == 0) red[wave] = vmax;
  __syncthreads();
  vmax = red[0];
#pragma unroll
  for (int j = 1; j < 8; ++j) vmax = fmaxf(vmax, red[j]);
  __syncthreads();

  float sum = 0.f;
#pragma unroll
  for (int j = 0; j < 8; ++j) {
    const float e = __expf(vals[j] - vmax);   // masked entries underflow to 0
    vals[j] = e;
    sum += e;
  }
#pragma unroll
  for (int off = 16; off > 0; off >>= 1) sum += __shfl_xor(sum, off, 32);
  if (lane == 0) red[wave] = sum;
  __syncthreads();
  sum = red[0];
#pragma unroll
  for (int j = 1; j < 8; ++j) sum += red[j];
  const float inv = 1.0f / sum;
#pragma unroll
  for (int j = 0; j < 8; ++j)
    s[threadIdx.x + j * 256] = (_Float16)(vals[j] * inv);
}

// out[row] = LN(X[row] (+ Y[row])) * g + b (biased var, eps=1e-5).
// Also writes f16 copy (out16) and, if pos!=null, f16 (res+pos) query input
// (xq16) for the next attention sublayer.  Safe when out aliases X.
__global__ __launch_bounds__(256)
void ln_residual(const float* __restrict__ X, const float* __restrict__ Y,
                 const float* __restrict__ g, const float* __restrict__ b,
                 float* __restrict__ out, _Float16* __restrict__ out16,
                 const float* __restrict__ pos, _Float16* __restrict__ xq16) {
  const int row = blockIdx.x;
  const int i   = threadIdx.x;
  const long long base = (long long)row * kD;
  float v = X[base + i];
  if (Y) v += Y[base + i];

  __shared__ float rs[8], rq[8];
  const int lane = threadIdx.x & 31;
  const int wave = threadIdx.x >> 5;
  float s = v, q = v * v;
#pragma unroll
  for (int off = 16; off > 0; off >>= 1) { s += __shfl_xor(s, off, 32); q += __shfl_xor(q, off, 32); }
  if (lane == 0) { rs[wave] = s; rq[wave] = q; }
  __syncthreads();
  s = 0.f; q = 0.f;
#pragma unroll
  for (int j = 0; j < 8; ++j) { s += rs[j]; q += rq[j]; }
  const float mean = s * (1.0f / kD);
  const float var  = q * (1.0f / kD) - mean * mean;
  const float res  = (v - mean) * rsqrtf(var + kLnEps) * g[i] + b[i];
  out[base + i] = res;
  if (out16) out16[base + i] = (_Float16)res;
  if (pos)   xq16[base + i]  = (_Float16)(res + pos[base + i]);
}

__global__ __launch_bounds__(256)
void add_f16(_Float16* __restrict__ o, const float* __restrict__ a,
             const float* __restrict__ b, long long n) {
  const long long i = (long long)blockIdx.x * blockDim.x + threadIdx.x;
  if (i < n) o[i] = (_Float16)(a[i] + b[i]);
}

__global__ __launch_bounds__(256)
void cvt_f16(_Float16* __restrict__ o, const float* __restrict__ a, long long n) {
  const long long i = (long long)blockIdx.x * blockDim.x + threadIdx.x;
  if (i < n) o[i] = (_Float16)a[i];
}

__global__ __launch_bounds__(256)
void copy_f32(float* __restrict__ o, const float* __restrict__ a, long long n) {
  const long long i = (long long)blockIdx.x * blockDim.x + threadIdx.x;
  if (i < n) o[i] = a[i];
}

// ---------------------------------------------------------------------------

static inline void run_gemm(hipStream_t st, const _Float16* A, const _Float16* W,
                            const float* bias, void* C,
                            int M, int Nn, int Kd, int lda, int ldb, int ldc,
                            long long aZ, long long bZ, long long cZ, int Z,
                            float alpha, int relu, int storeT, int outHalf,
                            int nWaveN) {
  dim3 grid(Nn / (32 * nWaveN), M / (32 * (8 / nWaveN)), Z);
  gemm_wmma_f16<<<grid, dim3(256), 0, st>>>(A, W, bias, C, M, Nn, Kd, lda, ldb,
                                            ldc, aZ, bZ, cZ, alpha, relu,
                                            storeT, outHalf, nWaveN);
}

extern "C" void kernel_launch(void* const* d_in, const int* in_sizes, int n_in,
                              void* d_out, int out_size, void* d_ws, size_t ws_size,
                              hipStream_t stream) {
  (void)in_sizes; (void)n_in; (void)out_size; (void)ws_size;
  const float* src     = (const float*)d_in[0];
  const float* tgt     = (const float*)d_in[1];
  const float* pos[2]  = {(const float*)d_in[2], (const float*)d_in[3]};
  const float* xyz[2]  = {(const float*)d_in[4], (const float*)d_in[5]};
  const float* sa_w    = (const float*)d_in[6];
  const float* sa_b    = (const float*)d_in[7];
  const float* sa_wo   = (const float*)d_in[8];
  const float* sa_bo   = (const float*)d_in[9];
  const float* ca_w    = (const float*)d_in[10];
  const float* ca_b    = (const float*)d_in[11];
  const float* ca_wo   = (const float*)d_in[12];
  const float* ca_bo   = (const float*)d_in[13];
  const float* w1      = (const float*)d_in[14];
  const float* b1      = (const float*)d_in[15];
  const float* w2      = (const float*)d_in[16];
  const float* b2      = (const float*)d_in[17];
  const float* ln_g    = (const float*)d_in[18];
  const float* ln_b    = (const float*)d_in[19];
  const float* fn_g    = (const float*)d_in[20];
  const float* fn_b    = (const float*)d_in[21];
  float* out = (float*)d_out;

  // ---- workspace carving ----
  float* ws = (float*)d_ws;
  float* cur[2] = {ws + 0 * kND, ws + 1 * kND};       // f32 residual stream
  float* tmp[2] = {ws + 2 * kND, ws + 3 * kND};       // f32 sublayer output
  _Float16* hp = (_Float16*)(ws + 4 * kND);
  _Float16* cur16[2] = {hp + 0 * kND, hp + 1 * kND};
  _Float16* xq16[2]  = {hp + 2 * kND, hp + 3 * kND};
  _Float16* Q16[2]   = {hp + 4 * kND, hp + 5 * kND};
  _Float16* K16[2]   = {hp + 6 * kND, hp + 7 * kND};
  _Float16* Vt16[2]  = {hp + 8 * kND, hp + 9 * kND};  // [D][N] head-transposed
  _Float16* att16[2] = {hp + 10 * kND, hp + 11 * kND};
  _Float16* hid16[2] = {hp + 12 * kND, hp + 12 * kND + kNDff};
  _Float16* wreg = hp + 12 * kND + 2 * kNDff;
  const long long nSaw = (long long)kL * 3 * kD * kD;   // 589824
  const long long nWo  = (long long)kL * kD * kD;       // 196608
  const long long nW1  = (long long)kL * kDff * kD;     // 786432
  _Float16* saw16  = wreg;               wreg += nSaw;
  _Float16* sawo16 = wreg;               wreg += nWo;
  _Float16* caw16  = wreg;               wreg += nSaw;
  _Float16* cawo16 = wreg;               wreg += nWo;
  _Float16* w1_16  = wreg;               wreg += nW1;
  _Float16* w2_16  = wreg;               wreg += nW1;
  _Float16* probs  = wreg;               // [H][N][N] f16, 64 MB, reused

  const float r2tab[kL] = {1.0f, 4.0f, 16.0f};      // radius^2 per layer
  const float inv_sqrt_hd = 0.1767766952966369f;    // 1/sqrt(32)
  const dim3 blk(256);
  const dim3 ew_grid((unsigned)(kND / 256));
  auto cvt_grid = [](long long n) { return dim3((unsigned)((n + 255) / 256)); };

  // ---- one-time f16 weight conversion ----
  cvt_f16<<<cvt_grid(nSaw), blk, 0, stream>>>(saw16,  sa_w,  nSaw);
  cvt_f16<<<cvt_grid(nWo),  blk, 0, stream>>>(sawo16, sa_wo, nWo);
  cvt_f16<<<cvt_grid(nSaw), blk, 0, stream>>>(caw16,  ca_w,  nSaw);
  cvt_f16<<<cvt_grid(nWo),  blk, 0, stream>>>(cawo16, ca_wo, nWo);
  cvt_f16<<<cvt_grid(nW1),  blk, 0, stream>>>(w1_16,  w1,    nW1);
  cvt_f16<<<cvt_grid(nW1),  blk, 0, stream>>>(w2_16,  w2,    nW1);

  copy_f32<<<ew_grid, blk, 0, stream>>>(cur[0], src, kND);
  copy_f32<<<ew_grid, blk, 0, stream>>>(cur[1], tgt, kND);
  cvt_f16<<<ew_grid, blk, 0, stream>>>(cur16[0], src, kND);
  cvt_f16<<<ew_grid, blk, 0, stream>>>(cur16[1], tgt, kND);
  // layer-0 SA query input: xq = src/tgt + pos (later layers come from LN)
  add_f16<<<ew_grid, blk, 0, stream>>>(xq16[0], cur[0], pos[0], kND);
  add_f16<<<ew_grid, blk, 0, stream>>>(xq16[1], cur[1], pos[1], kND);

  for (int i = 0; i < kL; ++i) {
    const _Float16* saw = saw16 + (long long)i * 3 * kD * kD;
    const _Float16* caw = caw16 + (long long)i * 3 * kD * kD;
    const float* sab = sa_b + (long long)i * 3 * kD;
    const float* cab = ca_b + (long long)i * 3 * kD;
    const float* lg  = ln_g + (long long)i * 3 * kD;
    const float* lb  = ln_b + (long long)i * 3 * kD;

    // ================= masked self-attention =================
    // (xq16 prepared by previous sublayer's LN, or the init adds for i==0)
    for (int c = 0; c < 2; ++c) {
      run_gemm(stream, xq16[c],  saw,              sab,          Q16[c], kN, kD, kD, kD, kD, kD, 0, 0, 0, 1, 1.f, 0, 0, 1, 2);
      run_gemm(stream, xq16[c],  saw + kD * kD,    sab + kD,     K16[c], kN, kD, kD, kD, kD, kD, 0, 0, 0, 1, 1.f, 0, 0, 1, 2);
      run_gemm(stream, cur16[c], saw + 2 * kD * kD, sab + 2 * kD, Vt16[c], kN, kD, kD, kD, kD, kN, 0, 0, 0, 1, 1.f, 0, 1, 1, 2);
    }
    for (int c = 0; c < 2; ++c) {
      run_gemm(stream, Q16[c], K16[c], nullptr, probs, kN, kN, kHd, kD, kD, kN,
               kHd, kHd, kNN, kH, inv_sqrt_hd, 0, 0, 1, 2);
      masked_softmax<<<dim3(kN, kH), blk, 0, stream>>>(probs, xyz[c], r2tab[i]);
      run_gemm(stream, probs, Vt16[c], nullptr, att16[c], kN, kHd, kN, kN, kN, kD,
               kNN, (long long)kHd * kN, kHd, kH, 1.f, 0, 0, 1, 1);
      run_gemm(stream, att16[c], sawo16 + (long long)i * kD * kD,
               sa_bo + (long long)i * kD, tmp[c], kN, kD, kD, kD, kD, kD,
               0, 0, 0, 1, 1.f, 0, 0, 0, 2);
      // LN + emit f16 state and xq16 = LN-out + pos for cross-attention
      ln_residual<<<dim3(kN), blk, 0, stream>>>(cur[c], tmp[c], lg, lb,
                                                cur[c], cur16[c], pos[c], xq16[c]);
    }

    // ================= dense cross-attention =================
    for (int c = 0; c < 2; ++c) {
      run_gemm(stream, xq16[c],  caw,              cab,          Q16[c], kN, kD, kD, kD, kD, kD, 0, 0, 0, 1, 1.f, 0, 0, 1, 2);
      run_gemm(stream, xq16[c],  caw + kD * kD,    cab + kD,     K16[c], kN, kD, kD, kD, kD, kD, 0, 0, 0, 1, 1.f, 0, 0, 1, 2);
      run_gemm(stream, cur16[c], caw + 2 * kD * kD, cab + 2 * kD, Vt16[c], kN, kD, kD, kD, kD, kN, 0, 0, 0, 1, 1.f, 0, 1, 1, 2);
    }
    for (int c = 0; c < 2; ++c) {
      const int o = 1 - c;  // cross: queries from c, keys/values from other
      run_gemm(stream, Q16[c], K16[o], nullptr, probs, kN, kN, kHd, kD, kD, kN,
               kHd, kHd, kNN, kH, inv_sqrt_hd, 0, 0, 1, 2);
      masked_softmax<<<dim3(kN, kH), blk, 0, stream>>>(probs, nullptr, 0.f);
      run_gemm(stream, probs, Vt16[o], nullptr, att16[c], kN, kHd, kN, kN, kN, kD,
               kNN, (long long)kHd * kN, kHd, kH, 1.f, 0, 0, 1, 1);
      run_gemm(stream, att16[c], cawo16 + (long long)i * kD * kD,
               ca_bo + (long long)i * kD, tmp[c], kN, kD, kD, kD, kD, kD,
               0, 0, 0, 1, 1.f, 0, 0, 0, 2);
      ln_residual<<<dim3(kN), blk, 0, stream>>>(cur[c], tmp[c], lg + kD, lb + kD,
                                                cur[c], cur16[c], nullptr, nullptr);
    }

    // ================= FFN =================
    for (int c = 0; c < 2; ++c) {
      run_gemm(stream, cur16[c], w1_16 + (long long)i * kDff * kD,
               b1 + (long long)i * kDff, hid16[c], kN, kDff, kD, kD, kD, kDff,
               0, 0, 0, 1, 1.f, /*relu=*/1, 0, 1, 2);
      run_gemm(stream, hid16[c], w2_16 + (long long)i * kD * kDff,
               b2 + (long long)i * kD, tmp[c], kN, kD, kDff, kDff, kDff, kD,
               0, 0, 0, 1, 1.f, 0, 0, 0, 2);
      // LN + emit xq16 for the next layer's self-attention
      ln_residual<<<dim3(kN), blk, 0, stream>>>(cur[c], tmp[c], lg + 2 * kD, lb + 2 * kD,
                                                cur[c], cur16[c], pos[c], xq16[c]);
    }
  }

  // final norm -> output (src then tgt, flat-concatenated)
  ln_residual<<<dim3(kN), blk, 0, stream>>>(cur[0], nullptr, fn_g, fn_b, out, nullptr, nullptr, nullptr);
  ln_residual<<<dim3(kN), blk, 0, stream>>>(cur[1], nullptr, fn_g, fn_b, out + kND, nullptr, nullptr, nullptr);
}
